// LSTM_86638080295450
// MI455X (gfx1250) — matrix operand order repaired and checked by reading, not compiled, not run on hardware
//
#include <hip/hip_runtime.h>

// ---------------------------------------------------------------------------
// LayerNorm-LSTM for MI455X (gfx1250, wave32, WMMA bf16 16x16x32, f32 acc)
// B=128, T=1024, D=256, U=512
// ---------------------------------------------------------------------------

typedef __attribute__((ext_vector_type(16))) __bf16 v16bf;
typedef __attribute__((ext_vector_type(8)))  __bf16 v8bf;
typedef __attribute__((ext_vector_type(8)))  float  v8f;
typedef __attribute__((address_space(3)))    __bf16 lds_bf16;

#define B_   128
#define T_   1024
#define D_   256
#define U_   512
#define HROW 520   // padded LDS row stride (bf16): 1040B = 65*16B -> conflict-free frag reads
#define APAD 264   // padded A-tile row stride (bf16): 528B = 33*16B -> conflict-free frag reads
#define TSTG 4     // timesteps staged per LDS pass in input_gemm

// ISA 7.12.2 operand K-mappings (16-bit, 16x16x32)
// A (16x32): lane holds row M=lane&15; element e -> K = e + (e&8) + 8*(lane>>4)
// B (32x16): lane holds col N=lane&15; element e -> K = e + (e&8) + 8*(lane>>4)
__device__ __forceinline__ int kmapB(int lane, int e) { return e + (e & 8) + ((lane >> 4) << 3); }

// ---------------------------------------------------------------------------
// Convert fp32 -> bf16 (x tensor)
// ---------------------------------------------------------------------------
__global__ void cvt_f32_bf16(const float* __restrict__ s, __bf16* __restrict__ d, long n) {
    for (long i = (long)blockIdx.x * blockDim.x + threadIdx.x; i < n;
         i += (long)gridDim.x * blockDim.x)
        d[i] = (__bf16)s[i];
}

// ---------------------------------------------------------------------------
// Pack 4 weight matrices [K x 512] into WMMA B-operand panels:
//   dst[g][kt][nt(32)][lane(32)][e(16)]  (each (nt,kt) panel = 1KB contiguous)
// so a lane's v16bf fragment is one 32-byte contiguous load.
// ---------------------------------------------------------------------------
__global__ void pack_b_panels(const float* __restrict__ s0, const float* __restrict__ s1,
                              const float* __restrict__ s2, const float* __restrict__ s3,
                              __bf16* __restrict__ dst, int ktiles) {
    long total = 4L * ktiles * 32 * 32 * 16;
    for (long idx = (long)blockIdx.x * blockDim.x + threadIdx.x; idx < total;
         idx += (long)gridDim.x * blockDim.x) {
        int e    = idx & 15;
        int lane = (idx >> 4) & 31;
        int nt   = (idx >> 9) & 31;
        int r    = (int)(idx >> 14);          // = g*ktiles + kt
        int kt   = r % ktiles;
        int g    = r / ktiles;
        const float* src = (g == 0) ? s0 : (g == 1) ? s1 : (g == 2) ? s2 : s3;
        int k = kt * 32 + kmapB(lane, e);
        int n = nt * 16 + (lane & 15);
        dst[idx] = (__bf16)src[(long)k * U_ + n];
    }
}

// ---------------------------------------------------------------------------
// Phase 1: input projections, stored in WMMA fragment layout:
//   xproj[t][g][btile][ntile][lane][8] (f32) = x @ W_g + b_g
// Block = (tchunk, btile, gate); 8 waves x 4 ntiles. The A-tile (16 batch rows
// x TSTG timesteps x 256) is shared by all 8 waves, so it is staged ONCE into
// LDS via the CDNA5 async global->LDS path (ASYNCcnt), then each wave reads
// its fragments from LDS. Cuts global x traffic 8x vs per-wave loads.
// ---------------------------------------------------------------------------
__global__ void __launch_bounds__(256) input_gemm(
        const __bf16* __restrict__ xb,      // x as bf16, [B][T][D]
        const __bf16* __restrict__ pW,      // packed W panels
        const float* __restrict__ bi, const float* __restrict__ bf_,
        const float* __restrict__ bc, const float* __restrict__ bo,
        float* __restrict__ xproj) {
    __shared__ __align__(16) __bf16 at[TSTG * 16 * APAD];   // ~33 KB

    const int tid    = threadIdx.x;
    const int lane   = tid & 31;
    const int wave   = tid >> 5;
    const int bid    = blockIdx.x;
    const int g      = bid & 3;
    const int btile  = (bid >> 2) & 7;
    const int tchunk = bid >> 5;

    const float* bptr = (g == 0) ? bi : (g == 1) ? bf_ : (g == 2) ? bc : bo;
    const int hsel = (lane >> 4) << 3;                 // A-operand K half-offset

    // 32-bit LDS byte offset of the staging tile (addrspacecast generic->local)
    const uint32_t lds_base = (uint32_t)(uintptr_t)(lds_bf16*)at;

    for (int tp = 0; tp < 16 / TSTG; ++tp) {
        const int t0 = tchunk * 16 + tp * TSTG;

        // ---- async-stage A tile: TSTG x 16 rows x 512B, 16B per thread-chunk
#pragma unroll
        for (int j = 0; j < (TSTG * 16 * 32) / 256; ++j) {   // 8 chunks/thread
            const int c    = tid + 256 * j;
            const int dgrp = c & 31;            // 16B group within a 512B row
            const int row  = (c >> 5) & 15;     // batch row within tile
            const int slot = c >> 9;            // timestep slot
            const uint32_t goff = (uint32_t)(
                (((uint32_t)(btile * 16 + row) * T_ + (uint32_t)(t0 + slot)) * D_
                 + dgrp * 8) * 2);
            const uint32_t loff = lds_base +
                (uint32_t)(((slot * 16 + row) * APAD + dgrp * 8) * 2);
            asm volatile("global_load_async_to_lds_b128 %0, %1, %2"
                         :: "v"(loff), "v"(goff), "s"(xb) : "memory");
        }
        asm volatile("s_wait_asynccnt 0x0" ::: "memory");
        __syncthreads();

        // ---- compute: each wave owns 4 ntiles, TSTG timesteps each --------
        for (int nn = 0; nn < 4; ++nn) {
            const int ntile = wave * 4 + nn;
            const float bias = bptr[ntile * 16 + (lane & 15)];
            v16bf bfrag[8];
#pragma unroll
            for (int kt = 0; kt < 8; ++kt)
                bfrag[kt] = *(const v16bf*)(pW +
                    (((long)(g * 8 + kt) * 32 + ntile) * 32 + lane) * 16);
#pragma unroll
            for (int slot = 0; slot < TSTG; ++slot) {
                v8f acc;
#pragma unroll
                for (int e = 0; e < 8; ++e) acc[e] = bias;
                const __bf16* arow = at + (slot * 16 + (lane & 15)) * APAD;
#pragma unroll
                for (int kt = 0; kt < 8; ++kt) {
                    const int k0 = kt * 32 + hsel;
                    union { v16bf v; v8bf h[2]; } a;
                    a.h[0] = *(const v8bf*)(arow + k0);
                    a.h[1] = *(const v8bf*)(arow + k0 + 16);
                    acc = __builtin_amdgcn_wmma_f32_16x16x32_bf16(
                        false, a.v, false, bfrag[kt], (short)0, acc, false, false);
                }
                const int t = t0 + slot;
                *(v8f*)(xproj + ((((long)t * 4 + g) * 8 + btile) * 32 + ntile) * 256
                        + lane * 8) = acc;
            }
        }
        __syncthreads();    // LDS reuse fence before next staging pass
    }
}

// ---------------------------------------------------------------------------
// Phase 2: persistent recurrence. grid = 8 blocks (one batch-tile of 16 rows
// each), block = 1024 threads = 32 waves; wave w owns output cols [16w,16w+16)
// for ALL 4 gates, so i/f/c~/o for a given (row,col) land in the same lane.
// h: bf16 in LDS (doubles as next step's A operand), c: registers.
// No cross-workgroup sync for the entire T=1024 loop. Next step's xproj
// fragments are prefetched (global_prefetch_b8) during current step's GEMM.
// ---------------------------------------------------------------------------
__global__ void __launch_bounds__(1024) recurrence(
        const float* __restrict__ xproj,
        const __bf16* __restrict__ pU,      // packed U panels (L2-resident, 2MB)
        const float* __restrict__ gamma, const float* __restrict__ beta,
        float* __restrict__ out) {
    __shared__ __align__(16) __bf16 hbuf[16 * HROW];
    __shared__ float psum[4][16][32];
    __shared__ float psq [4][16][32];
    __shared__ float smean[4][16];
    __shared__ float srstd[4][16];

    const int tid    = threadIdx.x;
    const int lane   = tid & 31;
    const int wave   = tid >> 5;            // == ntile
    const int btile  = blockIdx.x;
    const int col    = wave * 16 + (lane & 15);
    const int half   = lane >> 4;           // 0: rows 0..7 / 1: rows 8..15 in C/D frag

    const float g_l = gamma[col];
    const float b_l = beta[col];

    for (int i = tid; i < 16 * HROW; i += 1024) hbuf[i] = (__bf16)0.f;
    __syncthreads();

    v8f c;
#pragma unroll
    for (int e = 0; e < 8; ++e) c[e] = 0.f;

    for (int t = 0; t < T_; ++t) {
        // --- init accumulators with precomputed x-projection (+bias) -------
        v8f acc[4];
#pragma unroll
        for (int g = 0; g < 4; ++g) {
            const float* xp = xproj +
                ((((long)t * 4 + g) * 8 + btile) * 32 + wave) * 256 + lane * 8;
            acc[g] = *(const v8f*)xp;
        }
        // --- prefetch next step's fragments into cache during this GEMM ----
        if (t + 1 < T_) {
#pragma unroll
            for (int g = 0; g < 4; ++g)
                __builtin_prefetch(xproj +
                    ((((long)(t + 1) * 4 + g) * 8 + btile) * 32 + wave) * 256
                    + lane * 8, 0, 1);
        }
        // --- h @ U_g for 4 gates: A from LDS, B panels from L2-resident ws -
        for (int kt = 0; kt < 16; ++kt) {
            const int k0 = kt * 32 + (half << 3);
            const __bf16* ap = hbuf + (lane & 15) * HROW + k0;
            union { v16bf v; v8bf h[2]; } a;
            a.h[0] = *(const v8bf*)(ap);
            a.h[1] = *(const v8bf*)(ap + 16);
#pragma unroll
            for (int g = 0; g < 4; ++g) {
                const __bf16* bp = pU + (((long)(g * 16 + kt) * 32 + wave) * 32 + lane) * 16;
                v16bf bfr = *(const v16bf*)bp;
                acc[g] = __builtin_amdgcn_wmma_f32_16x16x32_bf16(
                    false, a.v, false, bfr, (short)0, acc[g], false, false);
            }
        }
        // --- LayerNorm partial sums: butterfly over the 16-lane half -------
#pragma unroll
        for (int g = 0; g < 4; ++g) {
#pragma unroll
            for (int e = 0; e < 8; ++e) {
                float s = acc[g][e];
                float q = s * s;
                for (int m = 1; m < 16; m <<= 1) {
                    s += __shfl_xor(s, m, 32);
                    q += __shfl_xor(q, m, 32);
                }
                if ((lane & 15) == 0) {
                    const int row = e + (half << 3);
                    psum[g][row][wave] = s;
                    psq [g][row][wave] = q;
                }
            }
        }
        __syncthreads();                    // also fences h reads vs h writes
        if (tid < 64) {                     // 64 (gate,row) statistics
            const int g = tid >> 4, row = tid & 15;
            float s = 0.f, q = 0.f;
            for (int w = 0; w < 32; ++w) { s += psum[g][row][w]; q += psq[g][row][w]; }
            const float mu  = s * (1.f / 512.f);
            const float var = q * (1.f / 512.f) - mu * mu;
            smean[g][row] = mu;
            srstd[g][row] = rsqrtf(var + 1e-3f);
        }
        __syncthreads();
        // --- normalize, activate, update c/h; h -> global f32 + LDS bf16 ---
#pragma unroll
        for (int e = 0; e < 8; ++e) {
            const int row = e + (half << 3);
            float ln[4];
#pragma unroll
            for (int g = 0; g < 4; ++g)
                ln[g] = (acc[g][e] - smean[g][row]) * srstd[g][row] * g_l + b_l;
            const float ig = 1.f / (1.f + __expf(-ln[0]));
            const float fg = 1.f / (1.f + __expf(-ln[1]));
            const float ct = tanhf(ln[2]);
            const float og = 1.f / (1.f + __expf(-ln[3]));
            const float cn = fg * c[e] + ig * ct;
            c[e] = cn;
            const float h = og * tanhf(cn);
            const int b = btile * 16 + row;
            out[((long)b * T_ + t) * U_ + col] = h;
            hbuf[row * HROW + col] = (__bf16)h;
        }
        __syncthreads();                    // h complete before next step's A
    }
}

// ---------------------------------------------------------------------------
extern "C" void kernel_launch(void* const* d_in, const int* in_sizes, int n_in,
                              void* d_out, int out_size, void* d_ws, size_t ws_size,
                              hipStream_t stream) {
    const float* x  = (const float*)d_in[0];
    const float* Wi = (const float*)d_in[1];
    const float* Wf = (const float*)d_in[2];
    const float* Wc = (const float*)d_in[3];
    const float* Wo = (const float*)d_in[4];
    const float* Ui = (const float*)d_in[5];
    const float* Uf = (const float*)d_in[6];
    const float* Uc = (const float*)d_in[7];
    const float* Uo = (const float*)d_in[8];
    const float* bi = (const float*)d_in[9];
    const float* bf_ = (const float*)d_in[10];
    const float* bc = (const float*)d_in[11];
    const float* bo = (const float*)d_in[12];
    const float* gamma = (const float*)d_in[13];
    const float* beta  = (const float*)d_in[14];
    float* out = (float*)d_out;

    char* ws = (char*)d_ws;
    __bf16* xb    = (__bf16*)(ws);                                   // 64 MiB
    __bf16* pW    = (__bf16*)(ws + (size_t)64 * 1024 * 1024);        //  1 MiB
    __bf16* pU    = (__bf16*)(ws + (size_t)65 * 1024 * 1024);        //  2 MiB
    float*  xproj = (float*)(ws + (size_t)68 * 1024 * 1024);         //  1 GiB

    cvt_f32_bf16<<<2048, 256, 0, stream>>>(x, xb, (long)B_ * T_ * D_);
    pack_b_panels<<<512, 256, 0, stream>>>(Wi, Wf, Wc, Wo, pW, 8);
    pack_b_panels<<<1024, 256, 0, stream>>>(Ui, Uf, Uc, Uo, pU, 16);
    // 64 tchunks * 8 btiles * 4 gates = 2048 blocks, 8 waves each
    input_gemm<<<2048, 256, 0, stream>>>(xb, pW, bi, bf_, bc, bo, xproj);
    recurrence<<<8, 1024, 0, stream>>>(xproj, pU, gamma, beta, out);
}
